// PatchWiseSpectralMixture2d_26225070309570
// MI455X (gfx1250) — matrix operand order repaired and checked by reading, not compiled, not run on hardware
//
#include <hip/hip_runtime.h>
#include <hip/hip_bf16.h>
#include <cstdint>

typedef __attribute__((ext_vector_type(16))) __bf16 v16bf;
typedef __attribute__((ext_vector_type(8)))  float  v8f;

#define HW      512
#define NPATCH  1024
#define XSTRIDE 264   // 256 + 8 bf16 pad: keeps ds_load_b128 16B-aligned, spreads banks

__device__ __forceinline__ uint16_t f2bf(float f) {
  uint32_t u = __float_as_uint(f);
  u += 0x7FFFu + ((u >> 16) & 1u);          // round-to-nearest-even
  return (uint16_t)(u >> 16);
}
__device__ __forceinline__ float bf2f(uint16_t h) {
  return __uint_as_float(((uint32_t)h) << 16);
}
__device__ __forceinline__ void split2(float v, uint16_t& hi, uint16_t& lo) {
  hi = f2bf(v);
  lo = f2bf(v - bf2f(hi));
}

// ---------------------------------------------------------------------------
// Kernel 1: per patch, build the real spatial kernel k_p = Re(IFFT2(S_p)).
// k_p[u,v] = (1/256) * sum_{a,b} S[a,b] * cos(2*pi*(u*a + v*b)/16)
// ---------------------------------------------------------------------------
__global__ __launch_bounds__(256) void spectral_kernel_build(
    const float* __restrict__ logits, const float* __restrict__ mu,
    const float* __restrict__ sigma, float* __restrict__ kg) {
  const int p = blockIdx.x;
  const int t = threadIdx.x;
  __shared__ float S[256];
  __shared__ float ct[16];
  if (t < 16) ct[t] = cosf((float)t * 0.39269908169872414f);  // 2*pi/16

  // softmax weights (redundant per-thread, trivial cost)
  float l0 = logits[p * 3 + 0], l1 = logits[p * 3 + 1], l2 = logits[p * 3 + 2];
  float mx = fmaxf(l0, fmaxf(l1, l2));
  float e0 = expf(l0 - mx), e1 = expf(l1 - mx), e2 = expf(l2 - mx);
  float inv = 1.0f / (e0 + e1 + e2);
  float w[3] = {e0 * inv, e1 * inv, e2 * inv};

  const int a = t >> 4, b = t & 15;
  float fy = (float)((a < 8) ? a : a - 16) * (1.0f / 16.0f);
  float fx = (float)((b < 8) ? b : b - 16) * (1.0f / 16.0f);
  float s = 1e-6f;  // JITTER
#pragma unroll
  for (int q = 0; q < 3; ++q) {
    float muy = mu[(p * 3 + q) * 2 + 0], mux = mu[(p * 3 + q) * 2 + 1];
    float sy = sigma[(p * 3 + q) * 2 + 0], sx = sigma[(p * 3 + q) * 2 + 1];
    float dy1 = (fy - muy) / sy, dx1 = (fx - mux) / sx;
    float dy2 = (fy + muy) / sy, dx2 = (fx + mux) / sx;
    s += w[q] * (expf(-0.5f * (dy1 * dy1 + dx1 * dx1)) +
                 expf(-0.5f * (dy2 * dy2 + dx2 * dx2)));
  }
  S[t] = s;
  __syncthreads();

  const int u = a, v = b;
  float acc = 0.0f;
  for (int aa = 0; aa < 16; ++aa) {
    const int ua = (u * aa) & 15;
    const float* Sr = &S[aa * 16];
#pragma unroll
    for (int bb = 0; bb < 16; ++bb) acc += Sr[bb] * ct[(ua + v * bb) & 15];
  }
  kg[p * 256 + t] = acc * (1.0f / 256.0f);
}

// ---------------------------------------------------------------------------
// Kernel 2: per patch, y[b,n] = sum_m x[b,m]*K[m,n] + bias[n] as split-bf16
// WMMA GEMM (M=64 batch, N=256 pixels, K=256 pixels), K via rotated k_p table.
// ---------------------------------------------------------------------------
__global__ __launch_bounds__(512) void spectral_gemm(
    const float* __restrict__ x, const float* __restrict__ bias,
    const float* __restrict__ kg, float* __restrict__ y) {
  const int p = blockIdx.x;
  const int ph = p >> 5, pw = p & 31;
  const int tid = threadIdx.x;
  const int lane = tid & 31, wave = tid >> 5;
  const int j = lane & 15, khalf = lane >> 4;
  const int mtile = wave & 1;      // which 16-batch tile inside the 32-batch pass
  const int ngroup = wave >> 1;    // 0..7 -> N-tiles {ngroup, ngroup+8}

  __shared__ uint16_t xhi[32 * XSTRIDE];
  __shared__ uint16_t xlo[32 * XSTRIDE];
  __shared__ uint16_t thi[16 * XSTRIDE];
  __shared__ uint16_t tlo[16 * XSTRIDE];

  // Rotated kernel table: Trot[j][r][e] = k_p[r][(j - e) & 15], split hi/lo.
  for (int idx = tid; idx < 4096; idx += 512) {
    int jj = idx >> 8, r = (idx >> 4) & 15, c = idx & 15;
    float v = kg[p * 256 + r * 16 + ((jj - c) & 15)];
    uint16_t h, l;
    split2(v, h, l);
    thi[jj * XSTRIDE + r * 16 + c] = h;
    tlo[jj * XSTRIDE + r * 16 + c] = l;
  }

  const float bias0 = bias[p * 256 + (ngroup)*16 + j];
  const float bias1 = bias[p * 256 + (ngroup + 8) * 16 + j];

  union FragB { v16bf v; uint4 q[2]; };
  union FragC { v8f v; float f[8]; };

  for (int pass = 0; pass < 2; ++pass) {
    __syncthreads();  // trot ready (pass 0) / previous pass done reading xhi/xlo
    // Stage 32 batches of this patch into LDS, pre-split hi/lo.
    for (int idx = tid; idx < 8192; idx += 512) {
      int bl = idx >> 8, m = idx & 255;
      int bg = pass * 32 + bl;
      float v = x[(size_t)bg * (HW * HW) + (size_t)(ph * 16 + (m >> 4)) * HW +
                  pw * 16 + (m & 15)];
      uint16_t h, l;
      split2(v, h, l);
      xhi[bl * XSTRIDE + m] = h;
      xlo[bl * XSTRIDE + m] = l;
    }
    __syncthreads();

    FragC acc0, acc1;
#pragma unroll
    for (int e = 0; e < 8; ++e) { acc0.f[e] = bias0; acc1.f[e] = bias1; }

#pragma unroll
    for (int kk = 0; kk < 8; ++kk) {
      // A fragment (16x32 bf16): row = batch (lane&15), per-lane contiguous K = 16*khalf + s.
      FragB ah, al;
      const int abase = (mtile * 16 + j) * XSTRIDE + kk * 32 + khalf * 16;
      ah.q[0] = *(const uint4*)&xhi[abase];
      ah.q[1] = *(const uint4*)&xhi[abase + 8];
      al.q[0] = *(const uint4*)&xlo[abase];
      al.q[1] = *(const uint4*)&xlo[abase + 8];
#pragma unroll
      for (int u2 = 0; u2 < 2; ++u2) {
        const int nt = ngroup + 8 * u2;
        const int r = (nt - kk * 2 - khalf) & 15;  // kernel row (i - i') & 15
        const int tb = j * XSTRIDE + r * 16;
        FragB bh, bl2;
        bh.q[0] = *(const uint4*)&thi[tb];
        bh.q[1] = *(const uint4*)&thi[tb + 8];
        bl2.q[0] = *(const uint4*)&tlo[tb];
        bl2.q[1] = *(const uint4*)&tlo[tb + 8];
        v8f c = (u2 == 0) ? acc0.v : acc1.v;
        // split-bf16 f32 emulation: hi*hi + lo*hi + hi*lo, f32 accumulate
        c = __builtin_amdgcn_wmma_f32_16x16x32_bf16(false, ah.v, false, bh.v,
                                                    (short)0, c, false, false);
        c = __builtin_amdgcn_wmma_f32_16x16x32_bf16(false, al.v, false, bh.v,
                                                    (short)0, c, false, false);
        c = __builtin_amdgcn_wmma_f32_16x16x32_bf16(false, ah.v, false, bl2.v,
                                                    (short)0, c, false, false);
        if (u2 == 0) acc0.v = c; else acc1.v = c;
      }
    }

    // Store: C/D layout -> M = v + 8*(lane>>4), N = lane&15.
#pragma unroll
    for (int u2 = 0; u2 < 2; ++u2) {
      const int nt = ngroup + 8 * u2;
      const FragC& acc = (u2 == 0) ? acc0 : acc1;
      const size_t obase = (size_t)(ph * 16 + nt) * HW + pw * 16 + j;
#pragma unroll
      for (int v = 0; v < 8; ++v) {
        int b = pass * 32 + mtile * 16 + v + 8 * khalf;
        y[(size_t)b * (HW * HW) + obase] = acc.f[v];
      }
    }
  }
}

extern "C" void kernel_launch(void* const* d_in, const int* in_sizes, int n_in,
                              void* d_out, int out_size, void* d_ws, size_t ws_size,
                              hipStream_t stream) {
  const float* x      = (const float*)d_in[0];
  const float* logits = (const float*)d_in[1];
  const float* mu     = (const float*)d_in[2];
  const float* sigma  = (const float*)d_in[3];
  const float* bias   = (const float*)d_in[4];
  float* y  = (float*)d_out;
  float* kg = (float*)d_ws;  // 1024 * 256 floats = 1 MB scratch

  spectral_kernel_build<<<NPATCH, 256, 0, stream>>>(logits, mu, sigma, kg);
  spectral_gemm<<<NPATCH, 512, 0, stream>>>(x, bias, kg, y);
}
